// EnSSM_77206332113401
// MI455X (gfx1250) — matrix operand order repaired and checked by reading, not compiled, not run on hardware
//
#include <hip/hip_runtime.h>
#include <hip/hip_bf16.h>

// ---------------- problem constants ----------------
#define B_      4
#define L_      2048
#define LP_     (L_ + 2)           // per-batch zero-padded length for conv3
#define CIN_    256
#define DM_     512
#define DI_     1024
#define DSTATE_ 16
#define DCONV_  4
#define DTR_    32
#define MROWS_  (B_ * L_)          // 8192 flattened (b,l) rows

typedef _Float16 f16;
typedef __attribute__((ext_vector_type(16))) _Float16 v16h;
typedef __attribute__((ext_vector_type(8)))  _Float16 v8h;
typedef __attribute__((ext_vector_type(8)))  float    v8f;
typedef __attribute__((ext_vector_type(4)))  int      v4i;
typedef __attribute__((ext_vector_type(8)))  int      v8i;
typedef __attribute__((ext_vector_type(4)))  unsigned int v4u;

// ---------------- gfx1250 async global->LDS path ----------------
#if defined(__AMDGCN__)
# if __has_builtin(__builtin_amdgcn_global_load_async_to_lds_b128)
#  define HAVE_ASYNC 1
# else
#  define HAVE_ASYNC 0
#  warning "gfx1250: __builtin_amdgcn_global_load_async_to_lds_b128 not available; falling back to sync LDS staging"
# endif
#else
# define HAVE_ASYNC 0
#endif

#if HAVE_ASYNC
typedef __attribute__((address_space(1))) v4i as1_v4i;   // global
typedef __attribute__((address_space(3))) v4i as3_v4i;   // LDS
#endif

// copy 16B global -> LDS (async on CDNA5: tracked by ASYNCcnt, no VGPR round trip)
__device__ inline void cp16_g2l(const f16* g, f16* l) {
#if HAVE_ASYNC
  __builtin_amdgcn_global_load_async_to_lds_b128(
      (as1_v4i*)(uintptr_t)g,
      (as3_v4i*)(unsigned int)(uintptr_t)l,
      0, 0);
#else
  *(v8h*)l = *(const v8h*)g;
#endif
}

__device__ inline void wait_async0() {
#if HAVE_ASYNC
# if __has_builtin(__builtin_amdgcn_s_wait_asynccnt)
  __builtin_amdgcn_s_wait_asynccnt(0);
# else
  asm volatile("s_wait_asynccnt 0x0" ::: "memory");
# endif
#endif
}

// ---------------- gfx1250 Tensor Data Mover path ----------------
#if defined(__AMDGCN__)
# if __has_builtin(__builtin_amdgcn_tensor_load_to_lds)
#  define HAVE_TDM 1
# else
#  define HAVE_TDM 0
#  warning "gfx1250: __builtin_amdgcn_tensor_load_to_lds not available; using async staging"
# endif
#else
# define HAVE_TDM 0
#endif

#if HAVE_TDM
// One TDM op loads a [tile_d1][tile_d0] f16 tile (row stride = stride_elems in
// memory) into LDS, inserting 4 DWORDs of pad after every 16 DWORDs so LDS rows
// land on a 40-f16 (80B) pitch. D# built per CDNA5 ISA 8.3/8.4.
__device__ inline void tdm_load_2d(const f16* gbase, unsigned lds_byte_addr,
                                   int tile_d0, int tile_d1, long stride_elems) {
  unsigned long long ga = (unsigned long long)(uintptr_t)gbase;
  v4u g0;
  g0.x = 1u;                                            // count=1 valid, user mode
  g0.y = lds_byte_addr;                                 // lds_addr[31:0]
  g0.z = (unsigned)ga;                                  // global_addr[31:0]
  g0.w = (unsigned)((ga >> 32) & 0x01FFFFFFull)         // global_addr[56:32]
         | (2u << 30);                                  // type = 2 ("image")
  const unsigned td0 = 0x40000000u, td1 = 0x40000000u;  // huge tensor dims: no OOB clip
  unsigned long long st = (unsigned long long)stride_elems;
  v8i g1;
  g1[0] = (int)((1u << 16)      // data_size = 1 -> 2 bytes
              | (1u << 20)      // pad_enable
              | (3u << 22)      // pad_interval: 2^(3+1)=16 DWORDs (=32 f16) per row
              | (3u << 25));    // pad_amount: 3+1=4 DWORDs (=8 f16)
  g1[1] = (int)((td0 & 0xFFFFu) << 16);                           // tensor_dim0[15:0]
  g1[2] = (int)(((td0 >> 16) & 0xFFFFu) | ((td1 & 0xFFFFu) << 16));
  g1[3] = (int)(((td1 >> 16) & 0xFFFFu) | (((unsigned)tile_d0 & 0xFFFFu) << 16));
  g1[4] = (int)((unsigned)tile_d1 & 0xFFFFu);                     // tile_dim1, tile_dim2=0
  g1[5] = (int)(unsigned)(st & 0xFFFFFFFFull);                    // dim0_stride[31:0]
  g1[6] = (int)(unsigned)((st >> 32) & 0xFFFFull);                // dim0_stride[47:32]
  g1[7] = 0;
  v4i gz = {0, 0, 0, 0};                                // groups 2/3: 2D tile, no iterate
# if __clang_major__ >= 23
  v8i gz8 = {0, 0, 0, 0, 0, 0, 0, 0};
  __builtin_amdgcn_tensor_load_to_lds(g0, g1, gz, gz, gz8, 0);
# else
  __builtin_amdgcn_tensor_load_to_lds(g0, g1, gz, gz, 0);
# endif
}

__device__ inline void wait_tensor0() {
# if __has_builtin(__builtin_amdgcn_s_wait_tensorcnt)
  __builtin_amdgcn_s_wait_tensorcnt(0);
# else
  asm volatile("s_wait_tensorcnt 0x0" ::: "memory");
# endif
}
#endif

// CDNA5 16-bit A-matrix (16x32) lane layout: lane = hf*16 + m;
// element e (0..15): K = (e<8 ? e : e+8) + hf*8  -> two contiguous runs of 8.
// B-matrix (32x16): lane = hf*16 + n; this code loads A and B both with element e: K = hf*16 + e.
// C/D f32 (16x16): vgpr r, lane j: M = r + (j/16)*8, N = j%16.
__device__ inline v16h cat8(v8h lo, v8h hi) {
  return __builtin_shufflevector(lo, hi, 0,1,2,3,4,5,6,7,8,9,10,11,12,13,14,15);
}

// ---------------- weight/activation prep: f32 -> f16 ----------------
__global__ __launch_bounds__(256) void cvt_f16(const float* __restrict__ s,
                                               f16* __restrict__ d, int n) {
  int i = blockIdx.x * 256 + threadIdx.x;
  if (i < n) d[i] = (f16)s[i];
}

// x [B][L][CIN] f32 -> x16 [B][L+2][CIN] f16 with zero guard rows per batch
// (so the conv3 +/-1 temporal shift never needs a boundary branch).
__global__ __launch_bounds__(256) void cvt_x16(const float* __restrict__ x,
                                               f16* __restrict__ x16) {
  int i = blockIdx.x * 256 + threadIdx.x;            // B*(L+2)*CIN total
  if (i >= B_ * LP_ * CIN_) return;
  int b = i / (LP_ * CIN_);
  int rem = i % (LP_ * CIN_);
  int lp = rem >> 8, c = rem & 255;
  f16 v = (f16)0.f;
  if (lp >= 1 && lp <= L_)
    v = (f16)x[((long)b * L_ + (lp - 1)) * CIN_ + c];
  x16[i] = v;
}

// conv1_w [512][256][3] -> w1p[kk][d][c]  ([3][DM][CIN], i.e. [N][K] per tap)
// conv_res_w [512][256]  -> wrt[d][c]     (already [N][K]; just convert)
__global__ __launch_bounds__(256) void prep_convw(const float* __restrict__ w1,
                                                  const float* __restrict__ wr,
                                                  f16* __restrict__ w1p,
                                                  f16* __restrict__ wrt) {
  int i = blockIdx.x * 256 + threadIdx.x;            // 4*512*256 total
  if (i >= 4 * DM_ * CIN_) return;
  int kk = i >> 17;                                  // 512*256 = 2^17
  int r  = i & (DM_ * CIN_ - 1);
  int d  = r >> 8, c = r & 255;
  if (kk < 3) w1p[i] = (f16)w1[(d * CIN_ + c) * 3 + kk];
  else        wrt[r] = (f16)wr[d * CIN_ + c];
}

// ---------------- generic WMMA GEMM: C[M,N] = A[M,K] * W[N,K]^T ----------------
// All-f16 inputs. block: 256 thr (8 waves). Block tile 128(M) x TNB(N), K-step 32.
// Waves 4(M) x 2(N); wave tile 32 x TNB/2. Double-buffered LDS, tiles staged by
// TDM (one tensor_load_to_lds per tile) or per-thread async b128 copies.
// epilogue: 0 = store, 1 = softplus(acc + bias[n]). Optional f16 shadow C16.
template <int TNB>
__global__ __launch_bounds__(256) void gemm_wmma_f16(
    const f16* __restrict__ A, int lda,
    const f16* __restrict__ W, int K,
    float* __restrict__ C, int ldc,
    f16* __restrict__ C16,
    const float* __restrict__ bias, int epilogue) {
  constexpr int JW = TNB / 32;                       // W-fragments per wave
  __shared__ __attribute__((aligned(16))) f16 As[2][128][40];
  __shared__ __attribute__((aligned(16))) f16 Ws[2][TNB][40];

  const int tid  = threadIdx.x;
  const int wave = tid >> 5, lane = tid & 31;
  const int wm = wave & 3, wn = wave >> 2;
  const int hf = lane >> 4, ln = lane & 15;
  const long Mbase = (long)blockIdx.y * 128;
  const int  Nbase = blockIdx.x * TNB;

  v8f acc[2][JW] = {};

  auto stage = [&](int k0, int buf) {
#if HAVE_TDM
    if (wave == 0) {                                 // A tile [128][32] via TDM
      tdm_load_2d(A + Mbase * (long)lda + k0,
                  (unsigned)(uintptr_t)&As[buf][0][0], 32, 128, lda);
    } else if (wave == 1) {                          // W tile [TNB][32] via TDM
      tdm_load_2d(W + (long)Nbase * K + k0,
                  (unsigned)(uintptr_t)&Ws[buf][0][0], 32, TNB, K);
    }
#else
    for (int c = tid; c < 512; c += 256) {           // A tile: 512 x 16B chunks
      int r = c >> 2, off = (c & 3) * 8;
      cp16_g2l(A + (Mbase + r) * (long)lda + k0 + off, &As[buf][r][off]);
    }
    for (int c = tid; c < TNB * 4; c += 256) {       // W tile: TNB*4 x 16B chunks
      int r = c >> 2, off = (c & 3) * 8;
      cp16_g2l(W + (long)(Nbase + r) * K + k0 + off, &Ws[buf][r][off]);
    }
#endif
  };
  auto wait_stage = [&]() {
#if HAVE_TDM
    if (wave < 2) wait_tensor0();
#else
    wait_async0();
#endif
  };

  stage(0, 0);
  wait_stage();
  __syncthreads();

  int cur = 0;
  for (int k0 = 0; k0 < K; k0 += 32) {
    const bool more = (k0 + 32 < K);                 // uniform
    if (more) stage(k0 + 32, cur ^ 1);               // overlap with WMMA below

    v16h af[2], wf[JW];
#pragma unroll
    for (int f = 0; f < 2; ++f) {
      int m = wm * 32 + f * 16 + ln;
      af[f] = cat8(*(const v8h*)&As[cur][m][hf * 16],
                   *(const v8h*)&As[cur][m][hf * 16 + 8]);
    }
#pragma unroll
    for (int j = 0; j < JW; ++j) {
      int n = wn * (TNB / 2) + j * 16 + ln;
      wf[j] = cat8(*(const v8h*)&Ws[cur][n][hf * 16],
                   *(const v8h*)&Ws[cur][n][hf * 16 + 8]);
    }
#pragma unroll
    for (int i = 0; i < 2; ++i)
#pragma unroll
      for (int j = 0; j < JW; ++j)
        acc[i][j] = __builtin_amdgcn_wmma_f32_16x16x32_f16(
            false, af[i], false, wf[j], (short)0, acc[i][j], false, false);

    if (more) wait_stage();
    __syncthreads();
    cur ^= 1;
  }

#pragma unroll
  for (int i = 0; i < 2; ++i)
#pragma unroll
    for (int j = 0; j < JW; ++j) {
      int col = Nbase + wn * (TNB / 2) + j * 16 + ln;
      float bv = (epilogue == 1) ? bias[col] : 0.f;
#pragma unroll
      for (int r = 0; r < 8; ++r) {
        long row = Mbase + wm * 32 + i * 16 + r + hf * 8;
        float v = acc[i][j][r];
        if (epilogue == 1) {                         // softplus(v + b)
          v += bv;
          v = (v > 20.f) ? v : log1pf(__expf(v));
        }
        C[row * (long)ldc + col] = v;
        if (C16) C16[row * (long)ldc + col] = (f16)v;
      }
    }
}

// ---------------- EnResBlock: u = relu(BN(conv3(x))) + conv1x1(x) ----------------
// block: 128 thr (4 waves), tile 16(l) x 64(d). K loop: kk(3) x c-chunks(8).
// All tiles (shifted x, conv3 W, res W) staged by TDM / async copies.
__global__ __launch_bounds__(128) void enres_wmma(
    const f16* __restrict__ x16,        // [B][L+2][CIN], zero guard rows
    const f16* __restrict__ w1p,        // [3][DM][CIN]
    const f16* __restrict__ wrt,        // [DM][CIN]
    const float* __restrict__ bn_g, const float* __restrict__ bn_b,
    const float* __restrict__ bn_mean, const float* __restrict__ bn_var,
    float* __restrict__ u,              // [B][L][DM]
    f16* __restrict__ u16) {            // f16 shadow for in_proj GEMM
  __shared__ __attribute__((aligned(16))) f16 As[16][40];
  __shared__ __attribute__((aligned(16))) f16 Ws[64][40];
  __shared__ __attribute__((aligned(16))) f16 Wr[64][40];

  const int tid = threadIdx.x;
  const int wave = tid >> 5, lane = tid & 31;
  const int hf = lane >> 4, ln = lane & 15;
  const int mt = blockIdx.y;                     // 0..511
  const int b  = mt >> 7;                        // 128 l-tiles per batch
  const int l0 = (mt & 127) << 4;
  const int n0 = blockIdx.x * 64;

  v8f acc1 = {}, acc2 = {};

  for (int kk = 0; kk < 3; ++kk) {
    for (int cb = 0; cb < CIN_; cb += 32) {
      __syncthreads();                             // prior tile fully consumed
#if HAVE_TDM
      // padded row lp = l0 + r + kk  <->  l = l0 + r + kk - 1 (guards give zeros)
      if (wave == 0)
        tdm_load_2d(x16 + ((long)(b * LP_ + l0 + kk)) * CIN_ + cb,
                    (unsigned)(uintptr_t)&As[0][0], 32, 16, CIN_);
      else if (wave == 1)
        tdm_load_2d(w1p + ((long)(kk * DM_ + n0)) * CIN_ + cb,
                    (unsigned)(uintptr_t)&Ws[0][0], 32, 64, CIN_);
      else if (wave == 2 && kk == 1)
        tdm_load_2d(wrt + (long)n0 * CIN_ + cb,
                    (unsigned)(uintptr_t)&Wr[0][0], 32, 64, CIN_);
      if (wave == 0 || wave == 1 || (wave == 2 && kk == 1)) wait_tensor0();
#else
      for (int c = tid; c < 64; c += 128) {        // x tile 16x32
        int r = c >> 2, off = (c & 3) * 8;
        cp16_g2l(x16 + ((long)(b * LP_ + l0 + kk + r)) * CIN_ + cb + off,
                 &As[r][off]);
      }
      for (int c = tid; c < 256; c += 128) {       // conv3 W tile 64x32
        int r = c >> 2, off = (c & 3) * 8;
        cp16_g2l(w1p + ((long)(kk * DM_ + n0 + r)) * CIN_ + cb + off,
                 &Ws[r][off]);
      }
      if (kk == 1)
        for (int c = tid; c < 256; c += 128) {     // res W tile 64x32
          int r = c >> 2, off = (c & 3) * 8;
          cp16_g2l(wrt + ((long)(n0 + r)) * CIN_ + cb + off, &Wr[r][off]);
        }
      wait_async0();
#endif
      __syncthreads();

      int m = ln;
      v16h af = cat8(*(const v8h*)&As[m][hf * 16], *(const v8h*)&As[m][hf * 16 + 8]);
      int n = wave * 16 + ln;
      v16h wfv = cat8(*(const v8h*)&Ws[n][hf * 16], *(const v8h*)&Ws[n][hf * 16 + 8]);
      acc1 = __builtin_amdgcn_wmma_f32_16x16x32_f16(
          false, af, false, wfv, (short)0, acc1, false, false);
      if (kk == 1) {
        v16h wfr = cat8(*(const v8h*)&Wr[n][hf * 16], *(const v8h*)&Wr[n][hf * 16 + 8]);
        acc2 = __builtin_amdgcn_wmma_f32_16x16x32_f16(
            false, af, false, wfr, (short)0, acc2, false, false);
      }
    }
  }

  int d = n0 + wave * 16 + ln;
  float sc = bn_g[d] * rsqrtf(bn_var[d] + 1e-5f);
  float mn = bn_mean[d], bb = bn_b[d];
#pragma unroll
  for (int r = 0; r < 8; ++r) {
    int l = l0 + r + hf * 8;
    float h = (acc1[r] - mn) * sc + bb;
    h = fmaxf(h, 0.f) + acc2[r];
    long idx = ((long)(b * L_ + l)) * DM_ + d;
    u[idx] = h;
    u16[idx] = (f16)h;
  }
}

// ---------------- depthwise causal conv (k=4) + SiLU ----------------
__global__ __launch_bounds__(256) void dwconv_silu(
    const float* __restrict__ xz,       // [M][2*DI], xm = cols [0,DI)
    const float* __restrict__ w,        // [DI][4]
    const float* __restrict__ bias,     // [DI]
    float* __restrict__ xs,             // [M][DI] f32 (for scan)
    f16* __restrict__ xs16) {           // f16 shadow (for x_proj GEMM)
  long gid = (long)blockIdx.x * 256 + threadIdx.x;
  if (gid >= (long)MROWS_ * DI_) return;
  int d = (int)(gid & (DI_ - 1));
  long bl = gid >> 10;
  int l = (int)(bl & (L_ - 1));
  long b = bl >> 11;
  float acc = bias[d];
#pragma unroll
  for (int j = 0; j < DCONV_; ++j) {
    int ls = l + j - (DCONV_ - 1);
    if (ls >= 0)
      acc += xz[((b * L_ + ls) * (2 * DI_)) + d] * w[d * DCONV_ + j];
  }
  float v = acc / (1.f + __expf(-acc));    // silu
  xs[gid] = v;
  xs16[gid] = (f16)v;
}

// ---------------- selective scan: 16 lanes (states) per channel ----------------
__global__ __launch_bounds__(256) void sel_scan(
    const float* __restrict__ dt,       // [M][DI]
    const float* __restrict__ dbl,      // [M][64]: [0,32)=dt_lo, [32,48)=B, [48,64)=C
    const float* __restrict__ xs,       // [M][DI]
    const float* __restrict__ xz,       // [M][2*DI], z = cols [DI,2*DI)
    const float* __restrict__ A_log,    // [DI][16]
    const float* __restrict__ Dp,       // [DI]
    f16* __restrict__ y16) {            // [M][DI] gated output (f16 for out_proj)
  long gid = (long)blockIdx.x * 256 + threadIdx.x;   // B*DI*16 = 65536 threads
  int n = (int)(gid & 15);
  long ch = gid >> 4;
  int d = (int)(ch & (DI_ - 1));
  int b = (int)(ch >> 10);
  float Aev = -__expf(A_log[d * DSTATE_ + n]);
  float Dv  = Dp[d];
  float h = 0.f;
  for (int t = 0; t < L_; ++t) {
    long base = (long)b * L_ + t;
    float dtv = dt[base * DI_ + d];
    float xv  = xs[base * DI_ + d];
    float Bv  = dbl[base * 64 + 32 + n];
    float Cv  = dbl[base * 64 + 48 + n];
    h = h * __expf(dtv * Aev) + dtv * Bv * xv;
    float p = h * Cv;
    p += __shfl_xor(p, 1);
    p += __shfl_xor(p, 2);
    p += __shfl_xor(p, 4);
    p += __shfl_xor(p, 8);
    if (n == 0) {
      float zv = xz[base * (2 * DI_) + DI_ + d];
      float g  = zv / (1.f + __expf(-zv));           // silu(z)
      y16[base * DI_ + d] = (f16)((p + Dv * xv) * g);
    }
  }
}

// ---------------- LayerNorm(m_out)*g + b + u ----------------
__global__ __launch_bounds__(256) void ln_residual(
    const float* __restrict__ mo, const float* __restrict__ u,
    const float* __restrict__ g, const float* __restrict__ bta,
    float* __restrict__ out) {
  int row = blockIdx.x, tid = threadIdx.x;
  __shared__ float part[8];
  __shared__ float stat[2];
  const float* r = mo + (long)row * DM_;
  float v0 = r[tid], v1 = r[tid + 256];
  float s = v0 + v1;
#pragma unroll
  for (int m = 1; m < 32; m <<= 1) s += __shfl_xor(s, m);
  if ((tid & 31) == 0) part[tid >> 5] = s;
  __syncthreads();
  if (tid == 0) {
    float t = 0.f;
    for (int i = 0; i < 8; ++i) t += part[i];
    stat[0] = t / (float)DM_;
  }
  __syncthreads();
  float mu = stat[0];
  float d0 = v0 - mu, d1 = v1 - mu;
  s = d0 * d0 + d1 * d1;
#pragma unroll
  for (int m = 1; m < 32; m <<= 1) s += __shfl_xor(s, m);
  __syncthreads();
  if ((tid & 31) == 0) part[tid >> 5] = s;
  __syncthreads();
  if (tid == 0) {
    float t = 0.f;
    for (int i = 0; i < 8; ++i) t += part[i];
    stat[1] = rsqrtf(t / (float)DM_ + 1e-6f);
  }
  __syncthreads();
  float rs = stat[1];
  long o = (long)row * DM_;
  out[o + tid]       = u[o + tid]       + d0 * rs * g[tid]       + bta[tid];
  out[o + tid + 256] = u[o + tid + 256] + d1 * rs * g[tid + 256] + bta[tid + 256];
}

// ---------------- launcher ----------------
extern "C" void kernel_launch(void* const* d_in, const int* in_sizes, int n_in,
                              void* d_out, int out_size, void* d_ws, size_t ws_size,
                              hipStream_t stream) {
  const float* x        = (const float*)d_in[0];
  const float* conv1_w  = (const float*)d_in[1];
  const float* convr_w  = (const float*)d_in[2];
  const float* bn_g     = (const float*)d_in[3];
  const float* bn_b     = (const float*)d_in[4];
  const float* bn_mean  = (const float*)d_in[5];
  const float* bn_var   = (const float*)d_in[6];
  const float* in_proj  = (const float*)d_in[7];
  const float* c1d_w    = (const float*)d_in[8];
  const float* c1d_b    = (const float*)d_in[9];
  const float* x_proj   = (const float*)d_in[10];
  const float* dt_proj  = (const float*)d_in[11];
  const float* dt_b     = (const float*)d_in[12];
  const float* A_log    = (const float*)d_in[13];
  const float* Dp       = (const float*)d_in[14];
  const float* out_proj = (const float*)d_in[15];
  const float* ln_g     = (const float*)d_in[16];
  const float* ln_b     = (const float*)d_in[17];
  float* out = (float*)d_out;

  // workspace carve-up
  char* p = (char*)d_ws;
  float* u     = (float*)p; p += (size_t)MROWS_ * DM_ * 4;
  float* xz    = (float*)p; p += (size_t)MROWS_ * 2 * DI_ * 4;
  float* xs    = (float*)p; p += (size_t)MROWS_ * DI_ * 4;
  float* dbl   = (float*)p; p += (size_t)MROWS_ * 64 * 4;
  float* dt    = (float*)p; p += (size_t)MROWS_ * DI_ * 4;
  float* mo    = (float*)p; p += (size_t)MROWS_ * DM_ * 4;
  f16* u16     = (f16*)p;   p += (size_t)MROWS_ * DM_ * 2;
  f16* xs16    = (f16*)p;   p += (size_t)MROWS_ * DI_ * 2;
  f16* dbl16   = (f16*)p;   p += (size_t)MROWS_ * 64 * 2;
  f16* y16     = (f16*)p;   p += (size_t)MROWS_ * DI_ * 2;
  f16* x16     = (f16*)p;   p += (size_t)B_ * LP_ * CIN_ * 2;
  f16* w_in    = (f16*)p;   p += (size_t)2 * DI_ * DM_ * 2;
  f16* w_x     = (f16*)p;   p += (size_t)64 * DI_ * 2;
  f16* w_dt    = (f16*)p;   p += (size_t)DI_ * DTR_ * 2;
  f16* w_out   = (f16*)p;   p += (size_t)DM_ * DI_ * 2;
  f16* w1p     = (f16*)p;   p += (size_t)3 * DM_ * CIN_ * 2;
  f16* wrt     = (f16*)p;   p += (size_t)DM_ * CIN_ * 2;

  // 1) prep -> f16 (weights + zero-padded input)
  cvt_f16<<<(2 * DI_ * DM_ + 255) / 256, 256, 0, stream>>>(in_proj, w_in, 2 * DI_ * DM_);
  cvt_f16<<<(64 * DI_ + 255) / 256, 256, 0, stream>>>(x_proj, w_x, 64 * DI_);
  cvt_f16<<<(DI_ * DTR_ + 255) / 256, 256, 0, stream>>>(dt_proj, w_dt, DI_ * DTR_);
  cvt_f16<<<(DM_ * DI_ + 255) / 256, 256, 0, stream>>>(out_proj, w_out, DM_ * DI_);
  prep_convw<<<(4 * DM_ * CIN_ + 255) / 256, 256, 0, stream>>>(conv1_w, convr_w, w1p, wrt);
  cvt_x16<<<(B_ * LP_ * CIN_ + 255) / 256, 256, 0, stream>>>(x, x16);

  // 2) EnResBlock -> u (f32) + u16 (f16)
  enres_wmma<<<dim3(DM_ / 64, MROWS_ / 16), 128, 0, stream>>>(
      x16, w1p, wrt, bn_g, bn_b, bn_mean, bn_var, u, u16);

  // 3) xz = u @ in_proj^T   [8192][2048]
  gemm_wmma_f16<128><<<dim3(2 * DI_ / 128, MROWS_ / 128), 256, 0, stream>>>(
      u16, DM_, w_in, DM_, xz, 2 * DI_, nullptr, nullptr, 0);

  // 4) xs = silu(depthwise_conv(xm) + b)
  dwconv_silu<<<(int)(((long)MROWS_ * DI_ + 255) / 256), 256, 0, stream>>>(
      xz, c1d_w, c1d_b, xs, xs16);

  // 5) dbl = xs @ x_proj^T  [8192][64] (+ f16 shadow for dt GEMM)
  gemm_wmma_f16<64><<<dim3(1, MROWS_ / 128), 256, 0, stream>>>(
      xs16, DI_, w_x, DI_, dbl, 64, dbl16, nullptr, 0);

  // 6) dt = softplus(dbl[:, :32] @ dt_proj^T + dt_b)  [8192][1024]
  gemm_wmma_f16<128><<<dim3(DI_ / 128, MROWS_ / 128), 256, 0, stream>>>(
      dbl16, 64, w_dt, DTR_, dt, DI_, nullptr, dt_b, 1);

  // 7) selective scan (fused +D*x and silu(z) gate) -> y16
  sel_scan<<<(B_ * DI_ * DSTATE_) / 256, 256, 0, stream>>>(
      dt, dbl, xs, xz, A_log, Dp, y16);

  // 8) m_out = y @ out_proj^T  [8192][512]
  gemm_wmma_f16<128><<<dim3(DM_ / 128, MROWS_ / 128), 256, 0, stream>>>(
      y16, DI_, w_out, DI_, mo, DM_, nullptr, nullptr, 0);

  // 9) out = u + LN(m_out)*g + b
  ln_residual<<<MROWS_, 256, 0, stream>>>(mo, u, ln_g, ln_b, out);
}
